// WrapperSSD_80041010528463
// MI455X (gfx1250) — compile-verified, matrix operand order checked by reading, not run on hardware
//
#include <hip/hip_runtime.h>
#include <hip/hip_bf16.h>
#include <stdint.h>

#define A_N     500000
#define C_N     91
#define NCLS    90
#define K_N     100
#define D_N     200
#define NC      9000          // NCLS*K_N candidates
#define NBINS   7200          // float-bit histogram bins for scores in (2^-7, 1]
#define CAP     2048          // per-class candidate capacity
#define SORTN   16384         // padded global sort size
#define NCHUNK  141           // ceil(NC/64)
#define TILE_A  128           // anchors per block in streaming passes
#define SCORE_T 0.01f
#define NMS_T   0.45f
#define NEG_INF -1e30f
#define BBOX_CLIP 4.135166556742356f   // log(1000/16)

typedef __attribute__((ext_vector_type(16))) _Float16     v16h;
typedef __attribute__((ext_vector_type(8)))  float        v8f;
typedef __attribute__((ext_vector_type(4)))  unsigned int v4u;
typedef __attribute__((ext_vector_type(8)))  int          v8i;
typedef __attribute__((ext_vector_type(4)))  int          v4i;

// ---------------------------------------------------------------------------
// TDM 1-D tile copy: global -> LDS, nElem 4-byte elements, issued by wave 0.
// D# per CDNA5 ISA 8.3/8.4: group0 = {count, lds_addr, global_addr, type=2},
// group1 = {data_size=4B, tensor_dim0=tile_dim0=nElem, dim1=1, stride0=nElem}.
// This toolchain exposes the 6-arg builtin (g0, g1, g2, g3, g4, cpol);
// groups 2/3/4 are zero for a <=2-D tensor descriptor.
// ---------------------------------------------------------------------------
__device__ __forceinline__ void tdm_load_tile(const float* src, float* lds, int nElem) {
#if __has_builtin(__builtin_amdgcn_tensor_load_to_lds)
  if (threadIdx.x < 32) {
    unsigned ldsAddr = (unsigned)(size_t)(void*)lds;     // generic LDS ptr low 32 = LDS byte addr
    unsigned long long ga = (unsigned long long)(size_t)src;
    unsigned nE = (unsigned)nElem;                       // <= 11648, fits tile_dim0 (16b)
    v4u g0;
    g0[0] = 1u;                                          // count=1 (valid user descriptor)
    g0[1] = ldsAddr;                                     // lds_addr (bytes)
    g0[2] = (unsigned)(ga & 0xFFFFFFFFu);                // global_addr[31:0]
    g0[3] = (unsigned)((ga >> 32) & 0x01FFFFFFu) | 0x80000000u; // ga[56:32] | type=2
    v8i g1;
    g1[0] = (int)(2u << 16);                             // workgroup_mask=0, data_size=4B
    g1[1] = (int)((nE & 0xFFFFu) << 16);                 // tensor_dim0[15:0]
    g1[2] = (int)(((nE >> 16) & 0xFFFFu) | (1u << 16));  // tensor_dim0[31:16], tensor_dim1=1
    g1[3] = (int)((nE & 0xFFFFu) << 16);                 // tile_dim0 = nElem
    g1[4] = 0;                                           // tile_dim1=0, tile_dim2=0 (1-D)
    g1[5] = (int)nE;                                     // tensor_dim0_stride lo32
    g1[6] = 0;
    g1[7] = 0;
    v4i z4; z4[0] = 0; z4[1] = 0; z4[2] = 0; z4[3] = 0;  // groups 2/3 unused (<=2D)
    v8i z8; z8[0] = 0; z8[1] = 0; z8[2] = 0; z8[3] = 0;
            z8[4] = 0; z8[5] = 0; z8[6] = 0; z8[7] = 0;  // extended group, unused
    __builtin_amdgcn_tensor_load_to_lds(g0, g1, z4, z4, z8, 0);
    __builtin_amdgcn_s_wait_tensorcnt(0);
  }
#else
  {
    const float4* s4 = (const float4*)src;               // nElem % 4 == 0 by construction
    float4* d4 = (float4*)lds;
    for (int i = threadIdx.x; i < (nElem >> 2); i += 256) d4[i] = s4[i];
  }
#endif
  __syncthreads();
}

__global__ void k_zero(unsigned* __restrict__ p, int n) {
  int i = blockIdx.x * blockDim.x + threadIdx.x;
  if (i < n) p[i] = 0u;
}

// ---------------------------------------------------------------------------
// Pass 1: softmax bias (rowmax + ln denom) per anchor, score histograms.
// 8 waves x 16 anchors; WMMA f16 ones-matmul does the 16-row exp-sum reduction.
// ---------------------------------------------------------------------------
__global__ void __launch_bounds__(256)
k_pass1(const float* __restrict__ logits, float* __restrict__ bias,
        unsigned* __restrict__ hist) {
  __shared__ float tile[TILE_A * C_N];   // 46592 B, stride 91 dwords (27 mod 64: conflict-free)
  __shared__ float wsum[TILE_A];
  const int tid  = threadIdx.x;
  const int base = blockIdx.x * TILE_A;
  const int nA   = min(TILE_A, A_N - base);
  if (tid == 0 && base + TILE_A < A_N)
    __builtin_prefetch(logits + (size_t)(base + TILE_A) * C_N, 0, 1); // warm L2 for next tile
  tdm_load_tile(logits + (size_t)base * C_N, tile, nA * C_N);

  const int lane = tid & 31;
  const int wv   = tid >> 5;
  const int m    = lane & 15;
  const int h    = lane >> 4;
  const int al   = wv * 16 + m;
  const bool act = (al < nA);
  float* row = &tile[al * C_N];

  // row max over this lane's class subset, combine halves
  float mx = -3.4e38f;
  for (int c = h; c < C_N; c += 2) mx = fmaxf(mx, row[c]);
  mx = fmaxf(mx, __shfl_xor(mx, 16));

  // exp in place
  for (int c = h; c < C_N; c += 2) row[c] = __expf(row[c] - mx);

  // WMMA row sums: D = Aexp(16x96, zero-padded) * ones(96x16), 3 chunks of K=32.
  // A layout (16-bit A 16x32): lane l holds M=l%16; elements 0..7 = K=kbase+0..7,
  // elements 8..15 = K=kbase+16..23, with kbase = kb*32 + 8*(l/16).
  // k0 = kbase+i <= 79 is always in-bounds; only k1 = kbase+16+i can reach 95,
  // so load from a clamped (in-bounds) index and zero via select — no EXEC branches.
  v16h bm;
#pragma unroll
  for (int i = 0; i < 16; ++i) bm[i] = (_Float16)1.0f;
  v8f acc = {};
#pragma unroll
  for (int kb = 0; kb < 3; ++kb) {
    const int kbase = kb * 32 + h * 8;
    v16h am;
#pragma unroll
    for (int i = 0; i < 8; ++i) {
      const int k0 = kbase + i;            // <= 79: always valid
      const int k1 = kbase + 16 + i;       // <= 95: may pad
      am[i] = (_Float16)row[k0];
      const float v1 = row[(k1 < C_N) ? k1 : 0];   // unconditional, in-bounds load
      am[i + 8] = (_Float16)((k1 < C_N) ? v1 : 0.0f);
    }
    acc = __builtin_amdgcn_wmma_f32_16x16x32_f16(false, am, false, bm,
                                                 (short)0, acc, false, false);
  }
  // D[M,N] identical across N; lane l vgpr v holds M = v + 8*(l>>4)
  if (m == 0) {
#pragma unroll
    for (int v = 0; v < 8; ++v) wsum[wv * 16 + h * 8 + v] = acc[v];
  }
  __syncthreads();
  const float denom = wsum[wv * 16 + m];
  const float inv   = 1.0f / denom;
  if (act && h == 0) bias[base + al] = mx + __logf(denom);

  // histogram of passing scores (classes 1..90), float-bit binning
  if (act) {
    for (int c = (h == 0 ? 2 : 1); c < C_N; c += 2) {
      float s = row[c] * inv;
      if (s > SCORE_T) {
        unsigned b = (__float_as_uint(s) - 0x3C000000u) >> 13;
        if (b > NBINS - 1) b = NBINS - 1;
        atomicAdd(&hist[(c - 1) * NBINS + b], 1u);
      }
    }
  }
}

// Per-class top-100 covering bin -> ln(score threshold)
__global__ void k_thresh(const unsigned* __restrict__ hist, float* __restrict__ lnThr) {
  int c = threadIdx.x;
  if (c >= NCLS) return;
  const unsigned* hrow = hist + (size_t)c * NBINS;
  unsigned cum = 0; int tb = 0;
  for (int b = NBINS - 1; b >= 0; --b) {
    cum += hrow[b];
    if (cum >= K_N) { tb = b; break; }
  }
  float lowScore = __uint_as_float(0x3C000000u + ((unsigned)tb << 13));
  lnThr[c] = __logf(fmaxf(lowScore, SCORE_T));
}

// ---------------------------------------------------------------------------
// Pass 2: log-domain compare only (no exp for rejects); append candidates.
// ---------------------------------------------------------------------------
__global__ void __launch_bounds__(256)
k_pass2(const float* __restrict__ logits, const float* __restrict__ bias,
        const float* __restrict__ lnThr, unsigned* __restrict__ clsCnt,
        unsigned long long* __restrict__ clsCand) {
  __shared__ float tile[TILE_A * C_N];
  __shared__ float sBias[TILE_A];
  __shared__ float sThr[NCLS];
  const int tid  = threadIdx.x;
  const int base = blockIdx.x * TILE_A;
  const int nA   = min(TILE_A, A_N - base);
  if (tid < NCLS) sThr[tid] = lnThr[tid];
  if (tid >= 128 && tid - 128 < nA) sBias[tid - 128] = bias[base + tid - 128];
  if (tid == 64 && base + TILE_A < A_N)
    __builtin_prefetch(logits + (size_t)(base + TILE_A) * C_N, 0, 1);
  tdm_load_tile(logits + (size_t)base * C_N, tile, nA * C_N);

  const int a = tid & 127;
  const int h = tid >> 7;
  if (a < nA) {
    const float bb = sBias[a];
    const float* row = &tile[a * C_N];
    const int anchorG = base + a;
    for (int c = 1 + h; c < C_N; c += 2) {
      float x = row[c];
      if (x > sThr[c - 1] + bb) {
        float s = __expf(x - bb);
        if (s > SCORE_T) {
          unsigned pos = atomicAdd(&clsCnt[c - 1], 1u);
          if (pos < CAP) {
            unsigned long long key =
                ((unsigned long long)__float_as_uint(s) << 32) | (unsigned)(~anchorG);
            clsCand[(size_t)(c - 1) * CAP + pos] = key;  // tie: lower anchor sorts first
          }
        }
      }
    }
  }
}

// Per-class bitonic top-100 + box decode/clip
__global__ void __launch_bounds__(256)
k_topk(const unsigned* __restrict__ clsCnt, const unsigned long long* __restrict__ clsCand,
       const float* __restrict__ bbox, const float* __restrict__ anchors,
       float* __restrict__ cScore, int* __restrict__ cAnchor,
       unsigned* __restrict__ cValid, float* __restrict__ cBox) {
  __shared__ unsigned long long sk[CAP];
  const int c = blockIdx.x, tid = threadIdx.x;
  const int cnt = min((int)clsCnt[c], CAP);
  for (int i = tid; i < CAP; i += 256)
    sk[i] = (i < cnt) ? clsCand[(size_t)c * CAP + i] : 0ull;
  __syncthreads();
  for (int k = 2; k <= CAP; k <<= 1)
    for (int j = k >> 1; j > 0; j >>= 1) {
      for (int p = tid; p < CAP; p += 256) {
        int q = p ^ j;
        if (q > p) {
          bool up = ((p & k) == 0);  // descending blocks
          unsigned long long x = sk[p], y = sk[q];
          if (up ? (x < y) : (x > y)) { sk[p] = y; sk[q] = x; }
        }
      }
      __syncthreads();
    }
  if (tid < K_N) {
    const int n = c * K_N + tid;
    unsigned long long key = sk[tid];
    float s = __uint_as_float((unsigned)(key >> 32));
    bool valid = (tid < cnt) && (s > SCORE_T);
    int anchor = valid ? (int)(~(unsigned)key) : 0;
    cScore[n]  = valid ? s : NEG_INF;
    cAnchor[n] = anchor;
    cValid[n]  = valid ? 1u : 0u;
    float b0 = 0.f, b1 = 0.f, b2 = 0.f, b3 = 0.f;
    if (valid) {
      float a0 = anchors[anchor*4+0], a1 = anchors[anchor*4+1];
      float a2 = anchors[anchor*4+2], a3 = anchors[anchor*4+3];
      float wa = a2 - a0, ha = a3 - a1;
      float cxa = a0 + 0.5f*wa, cya = a1 + 0.5f*ha;
      float dx = bbox[anchor*4+0]*0.1f, dy = bbox[anchor*4+1]*0.1f;
      float dw = fminf(bbox[anchor*4+2]*0.2f, BBOX_CLIP);
      float dh = fminf(bbox[anchor*4+3]*0.2f, BBOX_CLIP);
      float cx = dx*wa + cxa, cy = dy*ha + cya;
      float w = __expf(dw)*wa, hh = __expf(dh)*ha;
      b0 = fminf(fmaxf(cx - 0.5f*w , 0.f), 512.f);
      b1 = fminf(fmaxf(cy - 0.5f*hh, 0.f), 512.f);
      b2 = fminf(fmaxf(cx + 0.5f*w , 0.f), 512.f);
      b3 = fminf(fmaxf(cy + 0.5f*hh, 0.f), 512.f);
    }
    cBox[n*4+0]=b0; cBox[n*4+1]=b1; cBox[n*4+2]=b2; cBox[n*4+3]=b3;
  }
}

__global__ void __launch_bounds__(1024)
k_maxred(const float* __restrict__ cBox, float* __restrict__ maxC) {
  __shared__ float sm[1024];
  float v = 0.0f;
  for (int i = threadIdx.x; i < NC * 4; i += 1024) v = fmaxf(v, cBox[i]);
  sm[threadIdx.x] = v;
  __syncthreads();
  for (int s = 512; s > 0; s >>= 1) {
    if (threadIdx.x < s) sm[threadIdx.x] = fmaxf(sm[threadIdx.x], sm[threadIdx.x + s]);
    __syncthreads();
  }
  if (threadIdx.x == 0) *maxC = sm[0];
}

__device__ __forceinline__ unsigned monof(float f) {  // order-preserving float->u32
  unsigned u = __float_as_uint(f);
  return (u & 0x80000000u) ? ~u : (u | 0x80000000u);
}

__global__ void k_sortinit(const float* __restrict__ cScore,
                           unsigned long long* __restrict__ keys) {
  int p = blockIdx.x * blockDim.x + threadIdx.x;
  if (p >= SORTN) return;
  keys[p] = (p < NC) ? (((unsigned long long)monof(cScore[p]) << 32) | (unsigned)(~p))
                     : 0ull;   // padding sinks to the end of a descending sort
}

__global__ void k_bstep(unsigned long long* __restrict__ keys, int j, int k) {
  int p = blockIdx.x * blockDim.x + threadIdx.x;
  if (p >= SORTN) return;
  int q = p ^ j;
  if (q > p) {
    bool up = ((p & k) == 0);
    unsigned long long x = keys[p], y = keys[q];
    if (up ? (x < y) : (x > y)) { keys[p] = y; keys[q] = x; }
  }
}

__global__ void k_extract(const unsigned long long* __restrict__ keys,
                          const float* __restrict__ cBox, const unsigned* __restrict__ cValid,
                          const float* __restrict__ maxC, int* __restrict__ sortedN,
                          unsigned* __restrict__ sv, float* __restrict__ sbOff) {
  int p = blockIdx.x * blockDim.x + threadIdx.x;
  if (p >= NC) return;
  int n = (int)(~(unsigned)keys[p]);
  sortedN[p] = n;
  sv[p] = cValid[n];
  float off = (float)(n / K_N + 1) * (*maxC + 1.0f);   // per-class coordinate offset
  sbOff[p*4+0] = cBox[n*4+0] + off;
  sbOff[p*4+1] = cBox[n*4+1] + off;
  sbOff[p*4+2] = cBox[n*4+2] + off;
  sbOff[p*4+3] = cBox[n*4+3] + off;
}

__global__ void __launch_bounds__(64)
k_mask(const float* __restrict__ sbOff, unsigned long long* __restrict__ mask) {
  __shared__ float cb[64][4];
  const int t = threadIdx.x;
  const int j0 = blockIdx.x * 64;
  const int i  = blockIdx.y * 64 + t;
  int j = j0 + t;
  if (j < NC) { cb[t][0]=sbOff[j*4+0]; cb[t][1]=sbOff[j*4+1];
                cb[t][2]=sbOff[j*4+2]; cb[t][3]=sbOff[j*4+3]; }
  else        { cb[t][0]=0; cb[t][1]=0; cb[t][2]=0; cb[t][3]=0; }
  __syncthreads();
  if (i >= NC) return;
  float bi0=sbOff[i*4+0], bi1=sbOff[i*4+1], bi2=sbOff[i*4+2], bi3=sbOff[i*4+3];
  float areai = (bi2 - bi0) * (bi3 - bi1);
  unsigned long long bits = 0ull;
  for (int jj = 0; jj < 64; ++jj) {
    int jg = j0 + jj;
    if (jg <= i || jg >= NC) continue;                  // only later entries get suppressed
    float ltx = fmaxf(bi0, cb[jj][0]), lty = fmaxf(bi1, cb[jj][1]);
    float rbx = fminf(bi2, cb[jj][2]), rby = fminf(bi3, cb[jj][3]);
    float w = fmaxf(rbx - ltx, 0.f), hh = fmaxf(rby - lty, 0.f);
    float inter = w * hh;
    float areaj = (cb[jj][2]-cb[jj][0]) * (cb[jj][3]-cb[jj][1]);
    float iou = inter / fmaxf(areai + areaj - inter, 1e-9f);
    if (iou > NMS_T) bits |= (1ull << jj);
  }
  mask[(size_t)i * NCHUNK + blockIdx.x] = bits;
}

// Greedy scan, single wave32 in lockstep: no barriers for the 9000-step loop.
__global__ void __launch_bounds__(32)
k_scan(const unsigned long long* __restrict__ mask, const unsigned* __restrict__ sv,
       int* __restrict__ numKept, int* __restrict__ posG) {
  __shared__ unsigned long long removed_s[NCHUNK];
  volatile unsigned long long* removed = removed_s;      // cross-lane LDS, no compiler caching
  const int lane = threadIdx.x;
  for (int w = lane; w < NCHUNK; w += 32) removed[w] = 0ull;
  int k = 0;
  for (int i = 0; i < NC; ++i) {
    unsigned long long wbits = removed[i >> 6];          // uniform broadcast read
    bool keep = sv[i] && !((wbits >> (i & 63)) & 1ull);  // uniform across the wave
    if (keep) {
      if (k < D_N && lane == 0) posG[k] = i;
      ++k;
      for (int w = lane; w < NCHUNK; w += 32)            // disjoint words per lane
        removed[w] |= mask[(size_t)i * NCHUNK + w];
    }
  }
  if (lane == 0) *numKept = (k < D_N) ? k : D_N;
}

__global__ void __launch_bounds__(128)
k_out(const int* __restrict__ numKept, const int* __restrict__ posG,
      const int* __restrict__ sortedN, const float* __restrict__ cScore,
      const int* __restrict__ cAnchor, const float* __restrict__ cBox,
      const float* __restrict__ logits, float* __restrict__ out) {
  const int d = blockIdx.x, t = threadIdx.x;
  const int nk = *numKept;
  if (d < nk) {
    const int p = posG[d];
    const int n = sortedN[p];
    if (t == 0) {
      out[d*4+0] = cBox[n*4+1];          // [y1,x1,y2,x2] reorder
      out[d*4+1] = cBox[n*4+0];
      out[d*4+2] = cBox[n*4+3];
      out[d*4+3] = cBox[n*4+2];
      out[4*D_N + d] = cScore[n];
      out[5*D_N + d] = (float)(n / K_N + 1);
    }
    const int anchor = cAnchor[n];
    if (t < C_N) out[6*D_N + d*C_N + t] = logits[(size_t)anchor * C_N + t];
  } else {
    if (t == 0) {
      out[d*4+0]=0; out[d*4+1]=0; out[d*4+2]=0; out[d*4+3]=0;
      out[4*D_N + d]=0; out[5*D_N + d]=0;
    }
    if (t < C_N) out[6*D_N + d*C_N + t] = 0.f;
  }
}

extern "C" void kernel_launch(void* const* d_in, const int* in_sizes, int n_in,
                              void* d_out, int out_size, void* d_ws, size_t ws_size,
                              hipStream_t stream) {
  const float* logits  = (const float*)d_in[0];
  const float* bbox    = (const float*)d_in[1];
  const float* anchors = (const float*)d_in[2];
  float* out = (float*)d_out;
  char* ws = (char*)d_ws;
  (void)in_sizes; (void)n_in; (void)out_size; (void)ws_size;   // ws need ~17 MB

  size_t off = 0;
  auto alloc = [&](size_t bytes) { size_t o = off; off = (off + bytes + 255) & ~(size_t)255; return o; };
  float*               bias    = (float*)(ws + alloc((size_t)A_N * 4));
  unsigned*            hist    = (unsigned*)(ws + alloc((size_t)NCLS * NBINS * 4 + NCLS * 4));
  unsigned*            clsCnt  = hist + (size_t)NCLS * NBINS;   // contiguous with hist for zeroing
  float*               lnThr   = (float*)(ws + alloc(NCLS * 4));
  unsigned long long*  clsCand = (unsigned long long*)(ws + alloc((size_t)NCLS * CAP * 8));
  float*               cScore  = (float*)(ws + alloc(NC * 4));
  int*                 cAnchor = (int*)(ws + alloc(NC * 4));
  unsigned*            cValid  = (unsigned*)(ws + alloc(NC * 4));
  float*               cBox    = (float*)(ws + alloc(NC * 16));
  float*               maxC    = (float*)(ws + alloc(256));
  unsigned long long*  keys    = (unsigned long long*)(ws + alloc((size_t)SORTN * 8));
  int*                 sortedN = (int*)(ws + alloc(NC * 4));
  unsigned*            sv      = (unsigned*)(ws + alloc(NC * 4));
  float*               sbOff   = (float*)(ws + alloc(NC * 16));
  unsigned long long*  mask    = (unsigned long long*)(ws + alloc((size_t)NC * NCHUNK * 8));
  int*                 numKept = (int*)(ws + alloc(256));
  int*                 posG    = (int*)(ws + alloc(D_N * 4));

  const int zn = NCLS * NBINS + NCLS;
  k_zero<<<(zn + 255) / 256, 256, 0, stream>>>(hist, zn);
  const int gA = (A_N + TILE_A - 1) / TILE_A;
  k_pass1<<<gA, 256, 0, stream>>>(logits, bias, hist);
  k_thresh<<<1, 128, 0, stream>>>(hist, lnThr);
  k_pass2<<<gA, 256, 0, stream>>>(logits, bias, lnThr, clsCnt, clsCand);
  k_topk<<<NCLS, 256, 0, stream>>>(clsCnt, clsCand, bbox, anchors, cScore, cAnchor, cValid, cBox);
  k_maxred<<<1, 1024, 0, stream>>>(cBox, maxC);
  k_sortinit<<<SORTN / 256, 256, 0, stream>>>(cScore, keys);
  for (int k = 2; k <= SORTN; k <<= 1)
    for (int j = k >> 1; j > 0; j >>= 1)
      k_bstep<<<SORTN / 256, 256, 0, stream>>>(keys, j, k);
  k_extract<<<(NC + 255) / 256, 256, 0, stream>>>(keys, cBox, cValid, maxC, sortedN, sv, sbOff);
  k_mask<<<dim3(NCHUNK, NCHUNK), 64, 0, stream>>>(sbOff, mask);
  k_scan<<<1, 32, 0, stream>>>(mask, sv, numKept, posG);
  k_out<<<D_N, 128, 0, stream>>>(numKept, posG, sortedN, cScore, cAnchor, cBox, logits, out);
}
